// static_graph_constructor_65266323030680
// MI455X (gfx1250) — matrix operand order repaired and checked by reading, not compiled, not run on hardware
//
#include <hip/hip_runtime.h>

// Problem constants (from reference setup_inputs): B=4, C=16, N=4096, W=64.
#define B_ 4
#define C_ 16
#define N_ 4096
#define W_ 64
#define K_ 64   // conv output width 62, zero-padded to 64 for two K=32 WMMA steps

typedef _Float16 v16h __attribute__((ext_vector_type(16)));
typedef _Float16 v8h  __attribute__((ext_vector_type(8)));
typedef float    v8f  __attribute__((ext_vector_type(8)));

// ---------------------------------------------------------------------------
// Kernel 1: conv(1x3 over W, C->1) + bias, L2-normalize each row, split into
// f16 hi/lo pair (x = hi + lo, |lo| ~ 2^-12) and store zero-padded to K=64.
// One wave32 per (b,n) row; each lane covers w=lane and w=lane+32.
// Cost: reads 64 MB, writes 4 MB -> ~3 us, negligible.
// ---------------------------------------------------------------------------
__global__ __launch_bounds__(256) void prep_fn_kernel(
    const float* __restrict__ in, const float* __restrict__ cw,
    const float* __restrict__ cb, _Float16* __restrict__ hi,
    _Float16* __restrict__ lo)
{
    const int wave = threadIdx.x >> 5;
    const int lane = threadIdx.x & 31;
    const int row  = blockIdx.x * 8 + wave;   // [0, B_*N_)
    const int b    = row >> 12;               // /N_
    const int n    = row & (N_ - 1);

    const float* src = in + (((size_t)b * C_) * N_ + n) * W_;
    const float bias = cb[0];

    const int w1 = lane;        // 0..31  (always a valid conv position)
    const int w2 = lane + 32;   // 32..63 (valid if <= 61)

    float f1 = 0.f, f2 = 0.f;
#pragma unroll
    for (int c = 0; c < C_; ++c) {
        const float* r = src + (size_t)c * N_ * W_;
        const float k0 = cw[c * 3 + 0];
        const float k1 = cw[c * 3 + 1];
        const float k2 = cw[c * 3 + 2];
        f1 += r[w1] * k0 + r[w1 + 1] * k1 + r[w1 + 2] * k2;
        if (w2 <= 61) f2 += r[w2] * k0 + r[w2 + 1] * k1 + r[w2 + 2] * k2;
    }
    f1 += bias;
    f2 = (w2 <= 61) ? (f2 + bias) : 0.f;

    // wave32 butterfly reduction of sum of squares
    float ss = f1 * f1 + f2 * f2;
#pragma unroll
    for (int off = 16; off > 0; off >>= 1) ss += __shfl_xor(ss, off, 32);

    float nrm = sqrtf(ss);
    nrm = fmaxf(nrm, 1e-8f);

    const float g1 = f1 / nrm;
    const float g2 = f2 / nrm;

    const _Float16 h1 = (_Float16)g1;
    const _Float16 l1 = (_Float16)(g1 - (float)h1);
    const _Float16 h2 = (_Float16)g2;
    const _Float16 l2 = (_Float16)(g2 - (float)h2);

    const size_t o = (size_t)row * K_;
    hi[o + w1] = h1;  hi[o + w2] = h2;
    lo[o + w1] = l1;  lo[o + w2] = l2;
}

// ---------------------------------------------------------------------------
// Kernel 2: per-batch Gram matrix adj = fn . fn^T via v_wmma_f32_16x16x32_f16.
// Grid (32, 32, 4): one 128x128 f32 output tile per 256-thread block.
// 8 waves in a 2x4 layout; each wave computes a 64x32 sub-tile = eight 16x16
// accumulators, K=64 in two K=32 steps, 3 WMMAs per step (hi*hi, hi*lo, lo*hi)
// for ~f32 accuracy. fn buffers (4 MB) live in L2; the 256 MB result is
// streamed out with nontemporal stores (output-bandwidth-bound kernel).
// ---------------------------------------------------------------------------
__global__ __launch_bounds__(256) void gram_wmma_kernel(
    const _Float16* __restrict__ hi, const _Float16* __restrict__ lo,
    float* __restrict__ out)
{
    const int lane = threadIdx.x & 31;
    const int wave = threadIdx.x >> 5;
    const int b    = blockIdx.z;
    const int rowBase = blockIdx.y * 128 + (wave >> 2) * 64; // 2 row groups
    const int colBase = blockIdx.x * 128 + (wave & 3) * 32;  // 4 col groups
    const int m    = lane & 15;
    const int half = lane >> 4;
    const size_t base = (size_t)b * N_ * K_;

    v8f acc[4][2];
#pragma unroll
    for (int rt = 0; rt < 4; ++rt)
#pragma unroll
        for (int ct = 0; ct < 2; ++ct)
#pragma unroll
            for (int i = 0; i < 8; ++i) acc[rt][ct][i] = 0.f;

#pragma unroll
    for (int s = 0; s < 2; ++s) {            // K step: s*32 .. s*32+31
        // B fragments (K x N, N across lanes 0..15, K halves across lane halves):
        // lane holds B[kbase + 0..15][col] = fn[col][kbase + 0..15] -> one
        // contiguous 32B read per buffer.
        v16h bhi[2], blo[2];
#pragma unroll
        for (int ct = 0; ct < 2; ++ct) {
            const size_t g   = (size_t)(colBase + ct * 16 + m);
            const size_t off = base + g * K_ + s * 32 + half * 16;
            bhi[ct] = *(const v16h*)(hi + off);
            blo[ct] = *(const v16h*)(lo + off);
        }
#pragma unroll
        for (int rt = 0; rt < 4; ++rt) {
            // A fragments (M x K, row m per lane): lane needs
            // K = s*32 + half*16 + 0..15.
            const size_t g  = (size_t)(rowBase + rt * 16 + m);
            const size_t o0 = base + g * K_ + s * 32 + half * 16;
            v16h ahi = *(const v16h*)(hi + o0);
            v16h alo = *(const v16h*)(lo + o0);
#pragma unroll
            for (int ct = 0; ct < 2; ++ct) {
                acc[rt][ct] = __builtin_amdgcn_wmma_f32_16x16x32_f16(
                    false, ahi, false, bhi[ct], (short)0, acc[rt][ct], false, false);
                acc[rt][ct] = __builtin_amdgcn_wmma_f32_16x16x32_f16(
                    false, ahi, false, blo[ct], (short)0, acc[rt][ct], false, false);
                acc[rt][ct] = __builtin_amdgcn_wmma_f32_16x16x32_f16(
                    false, alo, false, bhi[ct], (short)0, acc[rt][ct], false, false);
            }
        }
    }

    // Store: C/D layout -> VGPR r holds row (rt*16 + half*8 + r), col = m.
    // 256 MB streamed once: use nontemporal stores to avoid polluting L2.
#pragma unroll
    for (int rt = 0; rt < 4; ++rt)
#pragma unroll
        for (int ct = 0; ct < 2; ++ct)
#pragma unroll
            for (int r = 0; r < 8; ++r) {
                const int row = rowBase + rt * 16 + half * 8 + r;
                const int col = colBase + ct * 16 + m;
                __builtin_nontemporal_store(
                    acc[rt][ct][r],
                    out + ((size_t)b * N_ + row) * N_ + col);
            }
}

// ---------------------------------------------------------------------------
// d_in[0] = inputs (4,16,4096,64) f32, d_in[1] = conv_w (48) f32,
// d_in[2] = conv_b (1) f32. d_out = adj (4,1,4096,4096) f32.
// d_ws: 4 MB used (fn_hi 2 MB + fn_lo 2 MB), fully rewritten every call.
// ---------------------------------------------------------------------------
extern "C" void kernel_launch(void* const* d_in, const int* in_sizes, int n_in,
                              void* d_out, int out_size, void* d_ws, size_t ws_size,
                              hipStream_t stream) {
    (void)in_sizes; (void)n_in; (void)out_size; (void)ws_size;
    const float* in = (const float*)d_in[0];
    const float* cw = (const float*)d_in[1];
    const float* cb = (const float*)d_in[2];
    float* out = (float*)d_out;

    _Float16* hi = (_Float16*)d_ws;
    _Float16* lo = hi + (size_t)B_ * N_ * K_;

    prep_fn_kernel<<<dim3((B_ * N_) / 8), 256, 0, stream>>>(in, cw, cb, hi, lo);
    gram_wmma_kernel<<<dim3(N_ / 128, N_ / 128, B_), 256, 0, stream>>>(hi, lo, out);
}
